// MultiHeadAttention_49100066127914
// MI455X (gfx1250) — matrix operand order repaired and checked by reading, not compiled, not run on hardware
//
#include <hip/hip_runtime.h>
#include <math.h>

// ---------------------------------------------------------------------------
// MHA forward (QKV+RoPE, causal flash attention, projection) for gfx1250.
// All matmuls on v_wmma_f32_16x16x32_bf16 (bf16 in, fp32 accumulate).
// GEMMs are register-blocked 64x64 per wave (4x4 WMMA tiles, 1:1 wmma:vmem).
// V is stored transposed [B,H,D,T] so attention PV fragments load as b128s.
// Epilogues use wave-uniform control only (no lane-predicated scatter).
// ---------------------------------------------------------------------------

typedef __bf16 bf16;
typedef bf16  __attribute__((ext_vector_type(16))) v16bf;
typedef bf16  __attribute__((ext_vector_type(8)))  v8bf;
typedef float __attribute__((ext_vector_type(8)))  v8f;
typedef unsigned int __attribute__((ext_vector_type(4))) u32x4;

#define NB    2
#define TSEQ  2048
#define CDIM  1024
#define NH    16
#define HD    64
#define MTOT  (NB * TSEQ)   // 4096 rows (batch*time)
#define NQKV  (3 * CDIM)    // 3072

// ---- bf16 helpers (RNE float->bf16, bf16->float) ---------------------------
static __device__ __forceinline__ unsigned short f2bf(float f) {
  unsigned int u = __builtin_bit_cast(unsigned int, f);
  u += 0x7FFFu + ((u >> 16) & 1u);
  return (unsigned short)(u >> 16);
}
static __device__ __forceinline__ float bf2f(unsigned short v) {
  unsigned int u = ((unsigned int)v) << 16;
  return __builtin_bit_cast(float, u);
}
// 16B vector load of 8 bf16
static __device__ __forceinline__ v8bf ld8(const unsigned short* p) {
  return __builtin_bit_cast(v8bf, *(const u32x4*)p);
}
static __device__ __forceinline__ v16bf mk16(v8bf lo, v8bf hi) {
  return __builtin_shufflevector(lo, hi, 0,1,2,3,4,5,6,7,8,9,10,11,12,13,14,15);
}
// Pack 8 accum floats into 8 bf16 (one b128 store's worth), element e -> t+e.
static __device__ __forceinline__ u32x4 pack8(v8f c) {
  u32x4 o;
#pragma unroll
  for (int w = 0; w < 4; ++w)
    o[w] = (unsigned)f2bf(c[2 * w]) | ((unsigned)f2bf(c[2 * w + 1]) << 16);
  return o;
}

// ---- kernel 0: fp32 -> bf16 conversion -------------------------------------
__global__ void f32_to_bf16_kernel(const float* __restrict__ src,
                                   unsigned short* __restrict__ dst, int n) {
  int i = blockIdx.x * blockDim.x + threadIdx.x;
  if (i < n) dst[i] = f2bf(src[i]);
}

// ---- 64x64 register-blocked WMMA GEMM: C = A[m0:+64,:K] * B[n0:+64,:K]^T ---
// A row-major [M,K], B row-major [N,K], K % 32 == 0.
// Fragment k order used here (contiguous; same for A and B):
//   A 16x32 bf16: lane row M = lane%16; elems {0..15}->K bkk..bkk+15,
//                 bkk = 16*(lane>=16)
//   B 32x16 bf16: lane col N = lane%16; elems {0..15}->K bkk..bkk+15,
//                 bkk = 16*(lane>=16)
static __device__ __forceinline__ void gemm_64x64_bf16(
    const unsigned short* __restrict__ A, const unsigned short* __restrict__ B,
    int m0, int n0, int K, v8f c[4][4]) {
  const int ln = threadIdx.x & 31, half = ln >> 4, l16 = ln & 15;
  const int akk = half * 8, bkk = half * 16;
  const unsigned short* ar[4];
  const unsigned short* br[4];
#pragma unroll
  for (int i = 0; i < 4; ++i) {
    ar[i] = A + (size_t)(m0 + i * 16 + l16) * K;
    br[i] = B + (size_t)(n0 + i * 16 + l16) * K;
  }
  for (int k = 0; k < K; k += 32) {
    v16bf af[4], bfr[4];
#pragma unroll
    for (int i = 0; i < 4; ++i) {
      af[i]  = mk16(ld8(ar[i] + k + bkk), ld8(ar[i] + k + bkk + 8));
      bfr[i] = mk16(ld8(br[i] + k + bkk), ld8(br[i] + k + bkk + 8));
    }
#pragma unroll
    for (int i = 0; i < 4; ++i)
#pragma unroll
      for (int j = 0; j < 4; ++j)
        c[i][j] = __builtin_amdgcn_wmma_f32_16x16x32_bf16(
            false, af[i], false, bfr[j], (short)0, c[i][j], false, false);
  }
}

// ---- kernel 1: QKV GEMM (64x64/wave), scatter into Q/K [B,H,T,D], Vt [B,H,D,T]
// The 64-wide N panel never crosses a q/k/v (1024) or head (64) boundary, and
// the 64-row M panel never crosses a batch boundary -> all control is uniform.
__global__ __launch_bounds__(256) void qkv_gemm_kernel(
    const unsigned short* __restrict__ xb, const unsigned short* __restrict__ wqkv,
    unsigned short* __restrict__ Qf, unsigned short* __restrict__ Kf,
    unsigned short* __restrict__ Vt) {
  const int w  = blockIdx.x * 8 + (threadIdx.x >> 5);
  const int tn = w % (NQKV / 64);             // consecutive waves share A panel
  const int tm = w / (NQKV / 64);
  const int m0 = tm * 64, n0 = tn * 64;
  v8f c[4][4];
  const v8f zero = {};
#pragma unroll
  for (int i = 0; i < 4; ++i)
#pragma unroll
    for (int j = 0; j < 4; ++j) c[i][j] = zero;
  gemm_64x64_bf16(xb, wqkv, m0, n0, CDIM, c);

  const int ln = threadIdx.x & 31, half = ln >> 4, l16 = ln & 15;
  const int sel   = n0 >> 10;           // uniform: 0=Q, 1=K, 2=V
  const int cc0   = n0 & 1023;          // uniform offset within q/k/v block
  const int bq    = m0 >> 11;           // uniform batch index
  const int tbase = m0 & (TSEQ - 1);    // uniform time base

  if (sel < 2) {
    // Q/K: [B,H,T,D]; per (i,r) row of 16 lanes stores 32B contiguous.
    unsigned short* qk = (sel == 0 ? Qf : Kf) +
        ((size_t)(bq * NH) * TSEQ + tbase + 8 * half) * HD + l16;
#pragma unroll
    for (int j = 0; j < 4; ++j) {
      const int h     = (cc0 + j * 16) >> 6;
      const int dbase = (cc0 + j * 16) & 63;
      unsigned short* dst = qk + (size_t)h * TSEQ * HD + dbase;
#pragma unroll
      for (int i = 0; i < 4; ++i)
#pragma unroll
        for (int r = 0; r < 8; ++r)
          dst[(i * 16 + r) * HD] = f2bf(c[i][j][r]);   // immediate offsets
    }
  } else {
    // V transposed: [B,H,D,T]; each lane's 8 rows are 8 consecutive t -> b128.
#pragma unroll
    for (int j = 0; j < 4; ++j) {
      const int h     = (cc0 + j * 16) >> 6;
      const int dbase = (cc0 + j * 16) & 63;
      unsigned short* dst = Vt +
          (((size_t)(bq * NH + h) * HD) + dbase + l16) * TSEQ + tbase + 8 * half;
#pragma unroll
      for (int i = 0; i < 4; ++i)
        *(u32x4*)(dst + i * 16) = pack8(c[i][j]);
    }
  }
}

// ---- kernel 2: RoPE on Q,K (in workspace), fold 1/sqrt(D) into Q -----------
__global__ void rope_kernel(unsigned short* __restrict__ Qf,
                            unsigned short* __restrict__ Kf) {
  const int idx = blockIdx.x * blockDim.x + threadIdx.x;  // [0, 32*2048*32)
  const int j  = idx & 31;                 // rotation pair index (d, d+32)
  const int t  = (idx >> 5) & (TSEQ - 1);
  const int bh = idx >> 16;
  const size_t base = ((size_t)bh * TSEQ + t) * HD;
  const float freq = __powf(10000.f, -(float)j * (1.f / 32.f));
  float sn, cs;
  __sincosf((float)t * freq, &sn, &cs);
  const float qs = 0.125f;                 // D^-0.5, folded into Q
  const float q0v = bf2f(Qf[base + j]), q1v = bf2f(Qf[base + j + 32]);
  Qf[base + j]      = f2bf((q0v * cs - q1v * sn) * qs);
  Qf[base + j + 32] = f2bf((q1v * cs + q0v * sn) * qs);
  const float k0v = bf2f(Kf[base + j]), k1v = bf2f(Kf[base + j + 32]);
  Kf[base + j]      = f2bf(k0v * cs - k1v * sn);
  Kf[base + j + 32] = f2bf(k1v * cs + k0v * sn);
}

// ---- kernel 3: causal flash attention, one wave per (b,h, 16-query tile) ---
// Q,K are [B,H,T,D]; V is transposed [B,H,D,T] so PV B-frags are contiguous.
__global__ __launch_bounds__(256) void flash_attn_kernel(
    const unsigned short* __restrict__ Q, const unsigned short* __restrict__ K,
    const unsigned short* __restrict__ Vt, unsigned short* __restrict__ O) {
  __shared__ __align__(16) unsigned short plds[8][16 * 32];  // P staging / wave
  const int wave = threadIdx.x >> 5;
  const int w  = blockIdx.x * 8 + wave;
  const int qt = w & ((TSEQ / 16) - 1);
  const int bh = w >> 7;                        // [0, B*H)
  const int q0 = qt * 16;
  const int ln = threadIdx.x & 31, half = ln >> 4, l16 = ln & 15;
  const unsigned short* Qb  = Q  + (size_t)bh * TSEQ * HD;
  const unsigned short* Kb  = K  + (size_t)bh * TSEQ * HD;
  const unsigned short* Vtb = Vt + (size_t)bh * HD * TSEQ;
  const int akk = half * 8, bkk = half * 16;

  // Q tile as two A-fragments (d=0..31, d=32..63); reused for all key chunks.
  const unsigned short* qrow = Qb + (size_t)(q0 + l16) * HD;
  const v16bf qa0 = mk16(ld8(qrow + bkk),      ld8(qrow + bkk + 8));
  const v16bf qa1 = mk16(ld8(qrow + 32 + bkk), ld8(qrow + 32 + bkk + 8));

  // Per-lane Vt row pointers for the 4 d-tiles (col d = j*16 + l16).
  const unsigned short* vr0 = Vtb + (size_t)(0  + l16) * TSEQ + bkk;
  const unsigned short* vr1 = Vtb + (size_t)(16 + l16) * TSEQ + bkk;
  const unsigned short* vr2 = Vtb + (size_t)(32 + l16) * TSEQ + bkk;
  const unsigned short* vr3 = Vtb + (size_t)(48 + l16) * TSEQ + bkk;

  float rmax[8], rsum[8];
  v8f acc0 = {}, acc1 = {}, acc2 = {}, acc3 = {};
#pragma unroll
  for (int r = 0; r < 8; ++r) { rmax[r] = -INFINITY; rsum[r] = 0.f; }

  unsigned short* pl = &plds[wave][0];

  for (int tk0 = 0; tk0 < q0 + 16; tk0 += 32) {        // uniform per wave
    // S = Q * K^T over two 16-key tiles; B frags are d-contiguous K rows.
    const unsigned short* kr0 = Kb + (size_t)(tk0 + l16) * HD;
    const unsigned short* kr1 = Kb + (size_t)(tk0 + 16 + l16) * HD;
    const v16bf kb00 = mk16(ld8(kr0 + bkk),      ld8(kr0 + bkk + 8));
    const v16bf kb01 = mk16(ld8(kr0 + 32 + bkk), ld8(kr0 + 32 + bkk + 8));
    const v16bf kb10 = mk16(ld8(kr1 + bkk),      ld8(kr1 + bkk + 8));
    const v16bf kb11 = mk16(ld8(kr1 + 32 + bkk), ld8(kr1 + 32 + bkk + 8));
    v8f s0 = {}, s1 = {};
    s0 = __builtin_amdgcn_wmma_f32_16x16x32_bf16(false, qa0, false, kb00, (short)0, s0, false, false);
    s0 = __builtin_amdgcn_wmma_f32_16x16x32_bf16(false, qa1, false, kb01, (short)0, s0, false, false);
    s1 = __builtin_amdgcn_wmma_f32_16x16x32_bf16(false, qa0, false, kb10, (short)0, s1, false, false);
    s1 = __builtin_amdgcn_wmma_f32_16x16x32_bf16(false, qa1, false, kb11, (short)0, s1, false, false);

    // Online softmax. Row m lives across the 16 lanes of one half-wave.
    const int key0 = tk0 + l16, key1 = key0 + 16;
#pragma unroll
    for (int r = 0; r < 8; ++r) {
      const int m = q0 + r + 8 * half;
      float x0 = (key0 <= m) ? s0[r] : -INFINITY;      // causal mask
      float x1 = (key1 <= m) ? s1[r] : -INFINITY;
      float cm = fmaxf(x0, x1);
#pragma unroll
      for (int off = 1; off < 16; off <<= 1) cm = fmaxf(cm, __shfl_xor(cm, off, 16));
      const float mnew = fmaxf(rmax[r], cm);
      const float fsc  = __expf(rmax[r] - mnew);       // 0 on first chunk
      const float p0 = __expf(x0 - mnew);
      const float p1 = __expf(x1 - mnew);
      float ps = p0 + p1;
#pragma unroll
      for (int off = 1; off < 16; off <<= 1) ps += __shfl_xor(ps, off, 16);
      rsum[r] = rsum[r] * fsc + ps;
      rmax[r] = mnew;
      acc0[r] *= fsc; acc1[r] *= fsc; acc2[r] *= fsc; acc3[r] *= fsc;
      const int mloc = r + 8 * half;                   // C-layout -> LDS rowmaj
      pl[mloc * 32 + l16]      = f2bf(p0);
      pl[mloc * 32 + 16 + l16] = f2bf(p1);
    }
    // Per-wave LDS ops are in-order: safe to read P back as an A fragment.
    const v16bf pa = mk16(ld8(pl + l16 * 32 + bkk), ld8(pl + l16 * 32 + bkk + 8));

    // V as four B fragments: contiguous 16B loads along keys from Vt rows.
    const v16bf vb0 = mk16(ld8(vr0 + tk0), ld8(vr0 + tk0 + 8));
    const v16bf vb1 = mk16(ld8(vr1 + tk0), ld8(vr1 + tk0 + 8));
    const v16bf vb2 = mk16(ld8(vr2 + tk0), ld8(vr2 + tk0 + 8));
    const v16bf vb3 = mk16(ld8(vr3 + tk0), ld8(vr3 + tk0 + 8));
    acc0 = __builtin_amdgcn_wmma_f32_16x16x32_bf16(false, pa, false, vb0, (short)0, acc0, false, false);
    acc1 = __builtin_amdgcn_wmma_f32_16x16x32_bf16(false, pa, false, vb1, (short)0, acc1, false, false);
    acc2 = __builtin_amdgcn_wmma_f32_16x16x32_bf16(false, pa, false, vb2, (short)0, acc2, false, false);
    acc3 = __builtin_amdgcn_wmma_f32_16x16x32_bf16(false, pa, false, vb3, (short)0, acc3, false, false);
  }

  // Normalize and write attention output as bf16 [B*T, C] for the proj GEMM.
  const int b = bh >> 4, h = bh & 15;
  unsigned short* orow =
      O + ((size_t)b * TSEQ + q0 + 8 * half) * CDIM + h * HD + l16;
#pragma unroll
  for (int r = 0; r < 8; ++r) {
    const float inv = 1.f / rsum[r];
    orow[(size_t)r * CDIM + 0]  = f2bf(acc0[r] * inv);
    orow[(size_t)r * CDIM + 16] = f2bf(acc1[r] * inv);
    orow[(size_t)r * CDIM + 32] = f2bf(acc2[r] * inv);
    orow[(size_t)r * CDIM + 48] = f2bf(acc3[r] * inv);
  }
}

// ---- kernel 4: projection GEMM (64x64/wave), fp32 result to d_out ----------
__global__ __launch_bounds__(256) void proj_gemm_kernel(
    const unsigned short* __restrict__ attn, const unsigned short* __restrict__ wproj,
    float* __restrict__ out) {
  const int w  = blockIdx.x * 8 + (threadIdx.x >> 5);
  const int tn = w % (CDIM / 64);
  const int tm = w / (CDIM / 64);
  const int m0 = tm * 64, n0 = tn * 64;
  v8f c[4][4];
  const v8f zero = {};
#pragma unroll
  for (int i = 0; i < 4; ++i)
#pragma unroll
    for (int j = 0; j < 4; ++j) c[i][j] = zero;
  gemm_64x64_bf16(attn, wproj, m0, n0, CDIM, c);

  const int ln = threadIdx.x & 31, half = ln >> 4, l16 = ln & 15;
  float* orow = out + ((size_t)m0 + 8 * half) * CDIM + n0 + l16;
#pragma unroll
  for (int i = 0; i < 4; ++i)
#pragma unroll
    for (int j = 0; j < 4; ++j)
#pragma unroll
      for (int r = 0; r < 8; ++r)
        orow[(size_t)(i * 16 + r) * CDIM + j * 16] = c[i][j][r];  // imm offsets
}

// ---------------------------------------------------------------------------
extern "C" void kernel_launch(void* const* d_in, const int* in_sizes, int n_in,
                              void* d_out, int out_size, void* d_ws, size_t ws_size,
                              hipStream_t stream) {
  const float* x     = (const float*)d_in[0];
  const float* Wqkv  = (const float*)d_in[1];
  const float* Wproj = (const float*)d_in[2];
  float* out = (float*)d_out;
  char* ws = (char*)d_ws;

  const size_t n_x     = (size_t)MTOT * CDIM;       // 4,194,304
  const size_t n_wqkv  = (size_t)NQKV * CDIM;       // 3,145,728
  const size_t n_wproj = (size_t)CDIM * CDIM;       // 1,048,576
  const size_t n_head  = (size_t)NB * NH * TSEQ * HD;

  unsigned short* xb     = (unsigned short*)(ws);
  unsigned short* wqkvb  = (unsigned short*)(ws + 2 * n_x);
  unsigned short* wprojb = (unsigned short*)(ws + 2 * (n_x + n_wqkv));
  unsigned short* Qf     = (unsigned short*)(ws + 2 * (n_x + n_wqkv + n_wproj));
  unsigned short* Kf     = Qf + n_head;
  unsigned short* Vt     = Kf + n_head;             // stored [B,H,D,T]
  unsigned short* attn   = Vt + n_head;             // total ~48 MB of ws

  f32_to_bf16_kernel<<<(int)((n_x + 255) / 256), 256, 0, stream>>>(x, xb, (int)n_x);
  f32_to_bf16_kernel<<<(int)((n_wqkv + 255) / 256), 256, 0, stream>>>(Wqkv, wqkvb, (int)n_wqkv);
  f32_to_bf16_kernel<<<(int)((n_wproj + 255) / 256), 256, 0, stream>>>(Wproj, wprojb, (int)n_wproj);

  qkv_gemm_kernel<<<(MTOT / 64) * (NQKV / 64) / 8, 256, 0, stream>>>(xb, wqkvb, Qf, Kf, Vt);
  rope_kernel<<<(NB * NH * TSEQ * 32) / 256, 256, 0, stream>>>(Qf, Kf);
  flash_attn_kernel<<<(NB * NH * (TSEQ / 16)) / 8, 256, 0, stream>>>(Qf, Kf, Vt, attn);
  proj_gemm_kernel<<<(MTOT / 64) * (CDIM / 64) / 8, 256, 0, stream>>>(attn, wprojb, out);
}
